// MultiHeadAttention_14963666059392
// MI455X (gfx1250) — matrix operand order repaired and checked by reading, not compiled, not run on hardware
//
#include <hip/hip_runtime.h>

// MHA forward for B=2,S=2048,F=1024,H=16,D=64 on gfx1250 (wave32, WMMA bf16,
// async-to-LDS staging). d_out = [out (2*2048*1024 f32)] ++ [attn (2*16*2048*2048 f32)].

typedef __attribute__((ext_vector_type(16))) __bf16 v16bf;
typedef __attribute__((ext_vector_type(8)))  __bf16 v8bf;
typedef __attribute__((ext_vector_type(8)))  float  v8f;

constexpr int Bc = 2, Sc = 2048, Fc = 1024, Hc = 16, Dc = 64;
constexpr int BM = 128, BN = 64, BK = 32;

// ---------- CDNA5 async copy helpers ----------

typedef __attribute__((address_space(3))) const void* lds_cptr;

__device__ __forceinline__ unsigned lds_off(const void* p) {
  // LDS-relative byte offset (AS3 pointers are 32-bit group-segment offsets)
  return (unsigned)(unsigned long long)(lds_cptr)p;
}

// 16B per-lane async copy global -> LDS (VGLOBAL GLOBAL_LOAD_ASYNC_TO_LDS_B128,
// tracked by ASYNCcnt).
__device__ __forceinline__ void async_ld_b128(unsigned ldsoff, const void* gptr) {
  asm volatile("global_load_async_to_lds_b128 %0, %1, off"
               :: "v"(ldsoff), "v"((unsigned long long)gptr) : "memory");
}

__device__ __forceinline__ void wait_async0() {
  asm volatile("s_wait_asynccnt 0x0" ::: "memory");
}

// ---------- WMMA fragment helpers (layouts per CDNA5 ISA 7.12.2) ----------

__device__ __forceinline__ v16bf frag16(v8bf lo, v8bf hi) {
  v16bf r;
#pragma unroll
  for (int i = 0; i < 8; ++i) { r[i] = lo[i]; r[i + 8] = hi[i]; }
  return r;
}

// A 16x32 bf16: lane L<16 -> M=L, K={0..7,16..23}; lane>=16 -> K={8..15,24..31}
__device__ __forceinline__ v16bf load_a_frag(const __bf16* tile, int lane) {
  const int row = lane & 15, kh = lane >> 4;
  const __bf16* p = tile + row * BK + kh * 8;
  return frag16(*(const v8bf*)p, *(const v8bf*)(p + 16));
}

// B 32x16 bf16 from LDS [n][k]: lane L<16 -> N=L, K={0..7,16..23}; lane>=16 -> K={8..15,24..31}
__device__ __forceinline__ v16bf load_b_frag(const __bf16* tile, int lane) {
  const int n = lane & 15, kg = lane >> 4;
  const __bf16* p = tile + n * BK + kg * 8;
  return frag16(*(const v8bf*)p, *(const v8bf*)(p + 16));
}

__device__ __forceinline__ v8f wmma_bf16(v16bf a, v16bf b, v8f c) {
  return __builtin_amdgcn_wmma_f32_16x16x32_bf16(false, a, false, b, (short)0, c,
                                                 false, false);
}

// ---------- fp32 -> bf16 convert ----------

__global__ __launch_bounds__(256) void cvt_f32_to_bf16(const float* __restrict__ in,
                                                       __bf16* __restrict__ out, int n) {
  int i = (blockIdx.x * 256 + threadIdx.x) * 4;
  if (i < n) {
#pragma unroll
    for (int j = 0; j < 4; ++j) out[i + j] = (__bf16)in[i + j];
  }
}

// ---------- generic WMMA GEMM: C = act(alpha * A @ Bw^T + bias) ----------
// A: [M,K] bf16 row-major; Bw: [N,K] bf16 row-major.
// Block tile 128x64, 8 waves, each wave 32x32 (4 WMMA accumulators).
// LDS double-buffered, filled with global_load_async_to_lds_b128.
// MODE 0: fp32 C[m,n] (batched via blockIdx.z strides)
// MODE 1: bf16 scatter to head-split [B,H,S,D]     (q,k projections)
// MODE 2: bf16 packed store to transposed heads [B,H,D,S]  (v projection)

template <int MODE, bool RELU>
__global__ __launch_bounds__(256) void wmma_gemm(
    const __bf16* __restrict__ A, const __bf16* __restrict__ Bw,
    const float* __restrict__ bias, float* __restrict__ Cf,
    __bf16* __restrict__ Cb, int M, int N, int K,
    long long sA, long long sB, long long sC, float alpha) {
  __shared__ __bf16 lA[2][BM * BK];
  __shared__ __bf16 lB[2][BN * BK];

  const int z = blockIdx.z;
  A += z * sA;
  Bw += z * sB;
  if (MODE == 0) Cf += z * sC;

  const int bm0 = blockIdx.x * BM;
  const int bn0 = blockIdx.y * BN;
  const int t = threadIdx.x;
  const int wave = t >> 5, lane = t & 31;
  const int wm = (wave & 3) * 32;   // wave M origin (4 waves over 128)
  const int wn = (wave >> 2) * 32;  // wave N origin (2 waves over 64)

  // staging: A = 128x32 bf16 (512 x 16B chunks, 2/thread), B = 64x32 (1/thread)
  const int ar0 = t >> 2, ac = (t & 3) * 8;  // rows 0..63
  const int ar1 = ar0 + 64;                  // rows 64..127
  const int br = t >> 2, bc = (t & 3) * 8;

  auto issue = [&](int buf, int k0) {
    async_ld_b128(lds_off(&lA[buf][ar0 * BK + ac]), &A[(long long)(bm0 + ar0) * K + k0 + ac]);
    async_ld_b128(lds_off(&lA[buf][ar1 * BK + ac]), &A[(long long)(bm0 + ar1) * K + k0 + ac]);
    async_ld_b128(lds_off(&lB[buf][br * BK + bc]), &Bw[(long long)(bn0 + br) * K + k0 + bc]);
  };

  v8f acc[2][2] = {};

  issue(0, 0);
  int cur = 0;
  for (int k0 = 0; k0 < K; k0 += BK) {
    wait_async0();
    __syncthreads();
    if (k0 + BK < K) issue(cur ^ 1, k0 + BK);

    v16bf a0 = load_a_frag(&lA[cur][wm * BK], lane);
    v16bf a1 = load_a_frag(&lA[cur][(wm + 16) * BK], lane);
    v16bf b0 = load_b_frag(&lB[cur][wn * BK], lane);
    v16bf b1 = load_b_frag(&lB[cur][(wn + 16) * BK], lane);
    acc[0][0] = wmma_bf16(a0, b0, acc[0][0]);
    acc[0][1] = wmma_bf16(a0, b1, acc[0][1]);
    acc[1][0] = wmma_bf16(a1, b0, acc[1][0]);
    acc[1][1] = wmma_bf16(a1, b1, acc[1][1]);
    cur ^= 1;
  }

  // C/D layout: lane -> N = lane&15, VGPR r -> M = 8*(lane>>4)+r
  const int ln = lane & 15, lh = lane >> 4;
#pragma unroll
  for (int mi = 0; mi < 2; ++mi) {
#pragma unroll
    for (int ni = 0; ni < 2; ++ni) {
      const int gm0 = bm0 + wm + mi * 16 + lh * 8;  // 8 consecutive rows from here
      const int gn = bn0 + wn + ni * 16 + ln;
      if (MODE == 2) {
        // packed transposed store: 8 consecutive s for fixed (h,d)
        const int bb = gm0 >> 11, s0 = gm0 & (Sc - 1);
        const int hh = gn >> 6, dd = gn & (Dc - 1);
        const float bv = bias[gn];
        v8bf pk;
#pragma unroll
        for (int r = 0; r < 8; ++r) pk[r] = (__bf16)(acc[mi][ni][r] + bv);
        *(v8bf*)&Cb[((((long long)bb * Hc + hh) * Dc) + dd) * Sc + s0] = pk;
      } else {
#pragma unroll
        for (int r = 0; r < 8; ++r) {
          const int gm = gm0 + r;
          float v = acc[mi][ni][r] * alpha;
          if (bias) v += bias[gn];
          if (RELU) v = fmaxf(v, 0.0f);
          if (MODE == 0) {
            Cf[(long long)gm * N + gn] = v;
          } else {  // MODE 1: [B,H,S,D] scatter
            const int bb = gm >> 11, ss = gm & (Sc - 1);
            const int hh = gn >> 6, dd = gn & (Dc - 1);
            Cb[((((long long)bb * Hc + hh) * Sc) + ss) * Dc + dd] = (__bf16)v;
          }
        }
      }
    }
  }
}

// ---------- row softmax over attn [BH*S rows, S cols], in place ----------

__global__ __launch_bounds__(256) void softmax_rows(float* __restrict__ attn) {
  float* p = attn + (long long)blockIdx.x * Sc;
  const int t = threadIdx.x;
  __shared__ float red[256];

  v8f vals = *(const v8f*)&p[t * 8];
  float m = vals[0];
#pragma unroll
  for (int i = 1; i < 8; ++i) m = fmaxf(m, vals[i]);
  red[t] = m;
  __syncthreads();
  for (int s = 128; s > 0; s >>= 1) {
    if (t < s) red[t] = fmaxf(red[t], red[t + s]);
    __syncthreads();
  }
  m = red[0];
  __syncthreads();

  float sum = 0.0f;
#pragma unroll
  for (int i = 0; i < 8; ++i) { vals[i] = __expf(vals[i] - m); sum += vals[i]; }
  red[t] = sum;
  __syncthreads();
  for (int s = 128; s > 0; s >>= 1) {
    if (t < s) red[t] += red[t + s];
    __syncthreads();
  }
  const float inv = 1.0f / red[0];
#pragma unroll
  for (int i = 0; i < 8; ++i) vals[i] *= inv;
  *(v8f*)&p[t * 8] = vals;
}

// ---------- ctx = attn(f32 -> bf16 on the fly) @ V, V^T staged async ----------
// attn: [S,S] f32 per (b,h); vhT: [B,H,D,S] bf16; out: merged heads [B,S,F] bf16.

__global__ __launch_bounds__(256) void wmma_ctx(const float* __restrict__ attn,
                                                const __bf16* __restrict__ vhT,
                                                __bf16* __restrict__ ctxb) {
  __shared__ __bf16 lA[BM * BK];
  __shared__ __bf16 lB[BN * BK];  // [d][k] -- direct rows of vhT

  const int z = blockIdx.z;  // b*H + h
  const float* Ap = attn + (long long)z * Sc * Sc;
  const __bf16* Vp = vhT + (long long)z * Dc * Sc;
  const int bm0 = blockIdx.x * BM;

  const int t = threadIdx.x;
  const int wave = t >> 5, lane = t & 31;
  const int wm = (wave & 3) * 32;
  const int wn = (wave >> 2) * 32;

  const int ar = t >> 1, ac = (t & 1) * 16;  // attn: 16 f32 per thread (one row half)
  const int br = t >> 2, bc = (t & 3) * 8;   // V^T: one 16B chunk per thread

  v8f acc[2][2] = {};

  for (int k0 = 0; k0 < Sc; k0 += BK) {
    // B tile: async contiguous copy of vhT rows
    async_ld_b128(lds_off(&lB[br * BK + bc]), &Vp[(long long)br * Sc + k0 + bc]);
    // A tile: fp32 attn -> bf16 in registers -> LDS
    v8f f0 = *(const v8f*)&Ap[(long long)(bm0 + ar) * Sc + k0 + ac];
    v8f f1 = *(const v8f*)&Ap[(long long)(bm0 + ar) * Sc + k0 + ac + 8];
    v8bf c0, c1;
#pragma unroll
    for (int i = 0; i < 8; ++i) { c0[i] = (__bf16)f0[i]; c1[i] = (__bf16)f1[i]; }
    *(v8bf*)&lA[ar * BK + ac] = c0;
    *(v8bf*)&lA[ar * BK + ac + 8] = c1;

    wait_async0();
    __syncthreads();

    v16bf a0 = load_a_frag(&lA[wm * BK], lane);
    v16bf a1 = load_a_frag(&lA[(wm + 16) * BK], lane);
    v16bf b0 = load_b_frag(&lB[wn * BK], lane);
    v16bf b1 = load_b_frag(&lB[(wn + 16) * BK], lane);
    acc[0][0] = wmma_bf16(a0, b0, acc[0][0]);
    acc[0][1] = wmma_bf16(a0, b1, acc[0][1]);
    acc[1][0] = wmma_bf16(a1, b0, acc[1][0]);
    acc[1][1] = wmma_bf16(a1, b1, acc[1][1]);
    __syncthreads();
  }

  const int bb = z >> 4, hh = z & 15;
  __bf16* cp = ctxb + (long long)bb * Sc * Fc + hh * Dc;  // [b, s, h*64 + d]
  const int ln = lane & 15, lh = lane >> 4;
#pragma unroll
  for (int mi = 0; mi < 2; ++mi) {
#pragma unroll
    for (int ni = 0; ni < 2; ++ni) {
#pragma unroll
      for (int r = 0; r < 8; ++r) {
        const int gm = bm0 + wm + mi * 16 + lh * 8 + r;  // s
        const int gn = wn + ni * 16 + ln;                // d
        cp[(long long)gm * Fc + gn] = (__bf16)acc[mi][ni][r];
      }
    }
  }
}

// ---------- launcher ----------

extern "C" void kernel_launch(void* const* d_in, const int* in_sizes, int n_in,
                              void* d_out, int out_size, void* d_ws, size_t ws_size,
                              hipStream_t stream) {
  const float* q    = (const float*)d_in[0];
  const float* k    = (const float*)d_in[1];
  const float* v    = (const float*)d_in[2];
  const float* wq_w = (const float*)d_in[3];
  const float* wq_b = (const float*)d_in[4];
  const float* wk_w = (const float*)d_in[5];
  const float* wk_b = (const float*)d_in[6];
  const float* wv_w = (const float*)d_in[7];
  const float* wv_b = (const float*)d_in[8];
  const float* fc_w = (const float*)d_in[9];
  const float* fc_b = (const float*)d_in[10];

  float* out  = (float*)d_out;
  float* attn = out + (long long)Bc * Sc * Fc;  // 4,194,304 floats in

  const long long MK = (long long)Bc * Sc * Fc;  // 4M elements
  const long long WK = (long long)Fc * Fc;       // 1M elements

  __bf16* qbf   = (__bf16*)d_ws;
  __bf16* kbf   = qbf + MK;
  __bf16* vbf   = kbf + MK;
  __bf16* wqbf  = vbf + MK;
  __bf16* wkbf  = wqbf + WK;
  __bf16* wvbf  = wkbf + WK;
  __bf16* fcbf  = wvbf + WK;
  __bf16* qh    = fcbf + WK;   // [B,H,S,D]
  __bf16* kh    = qh + MK;     // [B,H,S,D]
  __bf16* vhT   = kh + MK;     // [B,H,D,S]  (transposed for ctx B-tiles)
  __bf16* ctxbf = vhT + MK;    // [B,S,F]    (total ws use: 64 MB)

  const int M = Bc * Sc;  // 4096

  // 1) precision conversion
  {
    dim3 gA((MK / 4 + 255) / 256), gW((WK / 4 + 255) / 256);
    cvt_f32_to_bf16<<<gA, 256, 0, stream>>>(q, qbf, (int)MK);
    cvt_f32_to_bf16<<<gA, 256, 0, stream>>>(k, kbf, (int)MK);
    cvt_f32_to_bf16<<<gA, 256, 0, stream>>>(v, vbf, (int)MK);
    cvt_f32_to_bf16<<<gW, 256, 0, stream>>>(wq_w, wqbf, (int)WK);
    cvt_f32_to_bf16<<<gW, 256, 0, stream>>>(wk_w, wkbf, (int)WK);
    cvt_f32_to_bf16<<<gW, 256, 0, stream>>>(wv_w, wvbf, (int)WK);
    cvt_f32_to_bf16<<<gW, 256, 0, stream>>>(fc_w, fcbf, (int)WK);
  }

  // 2) projections (M=4096, N=1024, K=1024): q,k -> [B,H,S,D]; v -> [B,H,D,S]
  {
    dim3 g(M / BM, Fc / BN, 1);
    wmma_gemm<1, false><<<g, 256, 0, stream>>>(qbf, wqbf, wq_b, nullptr, qh, M, Fc, Fc,
                                               0, 0, 0, 1.0f);
    wmma_gemm<1, false><<<g, 256, 0, stream>>>(kbf, wkbf, wk_b, nullptr, kh, M, Fc, Fc,
                                               0, 0, 0, 1.0f);
    wmma_gemm<2, false><<<g, 256, 0, stream>>>(vbf, wvbf, wv_b, nullptr, vhT, M, Fc, Fc,
                                               0, 0, 0, 1.0f);
  }

  // 3) scores = qh @ kh^T * 1/sqrt(D), batched over B*H (raw f32 into attn)
  {
    dim3 g(Sc / BM, Sc / BN, Bc * Hc);
    wmma_gemm<0, false><<<g, 256, 0, stream>>>(qh, kh, nullptr, attn, nullptr, Sc, Sc, Dc,
                                               (long long)Sc * Dc, (long long)Sc * Dc,
                                               (long long)Sc * Sc, 0.125f);
  }

  // 4) softmax in place over last dim
  softmax_rows<<<Bc * Hc * Sc, 256, 0, stream>>>(attn);

  // 5) ctx = attn @ V -> merged-head bf16
  {
    dim3 g(Sc / BM, 1, Bc * Hc);
    wmma_ctx<<<g, 256, 0, stream>>>(attn, vhT, ctxbf);
  }

  // 6) out = relu(ctx @ fc^T + b) (M=4096, N=1024, K=1024)
  {
    dim3 g(M / BM, Fc / BN, 1);
    wmma_gemm<0, true><<<g, 256, 0, stream>>>(ctxbf, fcbf, fc_b, out, nullptr, M, Fc, Fc,
                                              0, 0, 0, 1.0f);
  }
}
